// GraphConvLayer_27015344292524
// MI455X (gfx1250) — matrix operand order repaired and checked by reading, not compiled, not run on hardware
//
#include <hip/hip_runtime.h>
#include <hip/hip_bf16.h>

typedef __attribute__((ext_vector_type(16))) __bf16 v16bf;
typedef __attribute__((ext_vector_type(8)))  float  v8f;

#define DD 128           // feature dim
#define BT 2             // B*T

union AFrag { v16bf v; unsigned short u[16]; };

static __device__ __forceinline__ unsigned short f2bf(float f) {
    unsigned u = __float_as_uint(f);
    unsigned r = u + 0x7FFFu + ((u >> 16) & 1u);   // round-to-nearest-even
    return (unsigned short)(r >> 16);
}

__global__ void zero_ws_kernel(float* __restrict__ p, size_t n) {
    size_t i = (size_t)blockIdx.x * blockDim.x + threadIdx.x;
    size_t stride = (size_t)gridDim.x * blockDim.x;
    for (; i < n; i += stride) p[i] = 0.0f;
}

// Persistent waves: each wave loops over 16-edge tiles (per t):
// msgs = relu(gather(neigh)·W_prep + b), scatter-added into summed[t,node_idx,:].
__global__ void __launch_bounds__(256)
edge_msg_kernel(const float* __restrict__ node, const int* __restrict__ edges,
                const float* __restrict__ Wp, const float* __restrict__ bp,
                float* __restrict__ summed, float* __restrict__ counts,
                int N, int E)
{
    __shared__ __align__(32) unsigned short WpT[DD * DD]; // bf16, [n][k]
    __shared__ float bprep[DD];
    for (int i = threadIdx.x; i < DD * DD; i += blockDim.x) {
        int k = i / DD, n = i % DD;            // W_prep row-major [k][n]
        WpT[n * DD + k] = f2bf(Wp[i]);
    }
    if (threadIdx.x < DD) bprep[threadIdx.x] = bp[threadIdx.x];
    __syncthreads();

    const int lane  = threadIdx.x & 31;
    const int wave  = threadIdx.x >> 5;
    const int nWaves = gridDim.x * (blockDim.x >> 5);
    const int tiles = (E + 15) >> 4;
    const int row  = lane & 15;
    const int half = lane >> 4;

    for (int wid = blockIdx.x * (blockDim.x >> 5) + wave;
         wid < tiles * BT; wid += nWaves) {
        const int tile = wid % tiles;
        const int t    = wid / tiles;
        const int e0   = tile * 16;

        if (t == 0 && half == 0) {              // per-edge count (once)
            int e = e0 + row;
            if (e < E) atomicAdd(&counts[edges[e]], 1.0f);
        }

        int eL = e0 + row; if (eL >= E) eL = E - 1;
        const int nidx = edges[E + eL];         // neighbor index (edges[1])
        const float* src = node + ((size_t)t * N + nidx) * DD;

        v8f acc[8];
        #pragma unroll
        for (int i = 0; i < 8; ++i) acc[i] = (v8f){0,0,0,0,0,0,0,0};

        #pragma unroll
        for (int ks = 0; ks < 4; ++ks) {
            const int k0 = ks * 32;
            const float4 f0 = *(const float4*)(src + k0 + half * 16);
            const float4 f1 = *(const float4*)(src + k0 + half * 16 + 4);
            const float4 f2 = *(const float4*)(src + k0 + half * 16 + 8);
            const float4 f3 = *(const float4*)(src + k0 + half * 16 + 12);
            AFrag a;
            a.u[0]=f2bf(f0.x); a.u[1]=f2bf(f0.y); a.u[2]=f2bf(f0.z); a.u[3]=f2bf(f0.w);
            a.u[4]=f2bf(f1.x); a.u[5]=f2bf(f1.y); a.u[6]=f2bf(f1.z); a.u[7]=f2bf(f1.w);
            a.u[8]=f2bf(f2.x); a.u[9]=f2bf(f2.y); a.u[10]=f2bf(f2.z); a.u[11]=f2bf(f2.w);
            a.u[12]=f2bf(f3.x); a.u[13]=f2bf(f3.y); a.u[14]=f2bf(f3.z); a.u[15]=f2bf(f3.w);
            #pragma unroll
            for (int nt = 0; nt < 8; ++nt) {
                const int n = nt * 16 + row;
                const v16bf bf = *(const v16bf*)&WpT[n * DD + k0 + half * 16];
                acc[nt] = __builtin_amdgcn_wmma_f32_16x16x32_bf16(
                    false, a.v, false, bf, (short)0, acc[nt], false, false);
            }
        }

        int dsts[8];
        #pragma unroll
        for (int v = 0; v < 8; ++v) {
            int e = e0 + v + 8 * half;          // C row M = v + 8*half
            dsts[v] = (e < E) ? edges[e] : -1;  // destination node (edges[0])
        }
        #pragma unroll
        for (int nt = 0; nt < 8; ++nt) {
            const int n = nt * 16 + row;
            const float bias = bprep[n];
            #pragma unroll
            for (int v = 0; v < 8; ++v) {
                if (dsts[v] >= 0) {
                    float val = acc[nt][v] + bias;
                    val = val > 0.0f ? val : 0.0f;  // relu
                    atomicAdd(&summed[((size_t)t * N + dsts[v]) * DD + n], val);
                }
            }
        }
    }
}

// Persistent waves: each wave loops over 16-node tiles (per t):
// out = relu([x ‖ summed/cnt]·W_upd + b)
__global__ void __launch_bounds__(256)
update_kernel(const float* __restrict__ node, const float* __restrict__ summed,
              const float* __restrict__ counts, const float* __restrict__ Wu,
              const float* __restrict__ bu, float* __restrict__ out, int N)
{
    __shared__ __align__(32) unsigned short WuT[DD * 2 * DD]; // bf16 [n][k], 64KB
    for (int i = threadIdx.x; i < 2 * DD * DD; i += blockDim.x) {
        int k = i / DD, n = i % DD;            // W_upd row-major [k][n], k<256
        WuT[n * (2 * DD) + k] = f2bf(Wu[i]);
    }
    __syncthreads();

    const int lane  = threadIdx.x & 31;
    const int wave  = threadIdx.x >> 5;
    const int nWaves = gridDim.x * (blockDim.x >> 5);
    const int tiles = (N + 15) >> 4;
    const int row  = lane & 15;
    const int half = lane >> 4;

    for (int wid = blockIdx.x * (blockDim.x >> 5) + wave;
         wid < tiles * BT; wid += nWaves) {
        const int tile = wid % tiles;
        const int t    = wid / tiles;

        int nd = tile * 16 + row; if (nd >= N) nd = N - 1;
        const float* xs = node   + ((size_t)t * N + nd) * DD;
        const float* ss = summed + ((size_t)t * N + nd) * DD;
        const float invc = 1.0f / fmaxf(counts[nd], 1.0f);

        v8f acc[8];
        #pragma unroll
        for (int i = 0; i < 8; ++i) acc[i] = (v8f){0,0,0,0,0,0,0,0};

        #pragma unroll
        for (int ks = 0; ks < 8; ++ks) {
            const int k0 = ks * 32;             // h-dim position, 0..255
            const float* s  = (ks < 4) ? xs : ss;
            const int    kb = (ks < 4) ? k0 : (k0 - DD);
            const float  sc = (ks < 4) ? 1.0f : invc;
            const float4 f0 = *(const float4*)(s + kb + half * 16);
            const float4 f1 = *(const float4*)(s + kb + half * 16 + 4);
            const float4 f2 = *(const float4*)(s + kb + half * 16 + 8);
            const float4 f3 = *(const float4*)(s + kb + half * 16 + 12);
            AFrag a;
            a.u[0]=f2bf(f0.x*sc); a.u[1]=f2bf(f0.y*sc); a.u[2]=f2bf(f0.z*sc); a.u[3]=f2bf(f0.w*sc);
            a.u[4]=f2bf(f1.x*sc); a.u[5]=f2bf(f1.y*sc); a.u[6]=f2bf(f1.z*sc); a.u[7]=f2bf(f1.w*sc);
            a.u[8]=f2bf(f2.x*sc); a.u[9]=f2bf(f2.y*sc); a.u[10]=f2bf(f2.z*sc); a.u[11]=f2bf(f2.w*sc);
            a.u[12]=f2bf(f3.x*sc); a.u[13]=f2bf(f3.y*sc); a.u[14]=f2bf(f3.z*sc); a.u[15]=f2bf(f3.w*sc);
            #pragma unroll
            for (int nt = 0; nt < 8; ++nt) {
                const int n = nt * 16 + row;
                const v16bf bf = *(const v16bf*)&WuT[n * (2 * DD) + k0 + half * 16];
                acc[nt] = __builtin_amdgcn_wmma_f32_16x16x32_bf16(
                    false, a.v, false, bf, (short)0, acc[nt], false, false);
            }
        }

        #pragma unroll
        for (int nt = 0; nt < 8; ++nt) {
            const int n = nt * 16 + row;
            const float bias = bu[n];
            #pragma unroll
            for (int v = 0; v < 8; ++v) {
                const int m = tile * 16 + v + 8 * half;  // output node row
                if (m < N) {
                    float val = acc[nt][v] + bias;
                    val = val > 0.0f ? val : 0.0f;
                    out[((size_t)t * N + m) * DD + n] = val;
                }
            }
        }
    }
}

extern "C" void kernel_launch(void* const* d_in, const int* in_sizes, int n_in,
                              void* d_out, int out_size, void* d_ws, size_t ws_size,
                              hipStream_t stream) {
    const float* node  = (const float*)d_in[0];   // [1,2,N,128] f32
    const int*   edges = (const int*)  d_in[1];   // [2,E] i32
    const float* Wp    = (const float*)d_in[2];   // [128,128]
    const float* bp    = (const float*)d_in[3];   // [128]
    const float* Wu    = (const float*)d_in[4];   // [256,128]
    const float* bu    = (const float*)d_in[5];   // [128]
    float* out = (float*)d_out;

    const int E = in_sizes[1] / 2;
    const int N = in_sizes[0] / (BT * DD);

    float* summed = (float*)d_ws;                       // BT*N*DD f32
    float* counts = summed + (size_t)BT * N * DD;       // N f32

    const size_t zn = (size_t)BT * N * DD + (size_t)N;
    zero_ws_kernel<<<1024, 256, 0, stream>>>(summed, zn);

    // persistent grids: amortize LDS weight staging over ~8 tiles per block (one per wave)
    const int tilesE  = (E + 15) / 16;
    const int wavesE  = tilesE * BT;
    int blocksE = (wavesE + 7) / 8;
    if (blocksE > 640) blocksE = 640;
    edge_msg_kernel<<<blocksE, 256, 0, stream>>>(node, edges, Wp, bp,
                                                 summed, counts, N, E);

    const int tilesN  = (N + 15) / 16;
    const int wavesU  = tilesN * BT;
    int blocksU = (wavesU + 7) / 8;
    if (blocksU > 320) blocksU = 320;
    update_kernel<<<blocksU, 256, 0, stream>>>(node, summed, counts,
                                               Wu, bu, out, N);
}
